// NDRSA_block_79482664780194
// MI455X (gfx1250) — matrix operand order repaired and checked by reading, not compiled, not run on hardware
//
#include <hip/hip_runtime.h>
#include <cstdint>

// ---------------------------------------------------------------------------
// Problem constants (match reference): B=4, L=2048, D=512, H=8, DH=64
// ---------------------------------------------------------------------------
constexpr int NB = 4;
constexpr int LQ = 2048;
constexpr int DD = 512;
constexpr int NH = 8;
constexpr int DHD = 64;

typedef __attribute__((ext_vector_type(16))) __bf16 v16bf;
typedef __attribute__((ext_vector_type(8)))  float   v8f;
typedef __attribute__((ext_vector_type(8)))  __bf16  bf16x8;

static __device__ __forceinline__ v8f vzero8() {
  v8f v;
#pragma unroll
  for (int i = 0; i < 8; ++i) v[i] = 0.0f;
  return v;
}

// B-fragment (32x16, K-major contiguous): element e -> K = base + e, N = lane%16.
// Caller points p at BT[n]*K + kb + (hi?16:0); loads 16 contiguous bf16 (32B).
static __device__ __forceinline__ v16bf ld_frag_contig(const __bf16* p) {
  bf16x8 x0 = *(const bf16x8*)(p);
  bf16x8 x1 = *(const bf16x8*)(p + 8);
  v16bf r;
#pragma unroll
  for (int e = 0; e < 8; ++e) { r[e] = x0[e]; r[e + 8] = x1[e]; }
  return r;
}

// A-fragment (16x32 row-major source): lane lo holds row M=lo.
// element e -> K = 2*hi8 + e (contiguous). Two 16B loads.
static __device__ __forceinline__ v16bf ld_afrag(const __bf16* rowp, int hi8) {
  bf16x8 x0 = *(const bf16x8*)(rowp + 2 * hi8);
  bf16x8 x1 = *(const bf16x8*)(rowp + 2 * hi8 + 8);
  v16bf r;
#pragma unroll
  for (int e = 0; e < 8; ++e) { r[e] = x0[e]; r[e + 8] = x1[e]; }
  return r;
}

static __device__ __forceinline__ v8f wmma_bf16(v16bf a, v16bf b, v8f c) {
  return __builtin_amdgcn_wmma_f32_16x16x32_bf16(false, a, false, b, (short)0, c,
                                                 false, false);
}

// ---------------------------------------------------------------------------
// Generic bf16 WMMA GEMM: C[M][N] = A[M][K] * BT[N][K]^T + bias[N]
// One wave per block computes a 32x64 tile (8 accumulators, B reused 2x).
// B fragments are preloaded as an array so loads pipeline against WMMAs.
// ---------------------------------------------------------------------------
enum { EPI_F32 = 0, EPI_HEAD = 1, EPI_HEADT = 2, EPI_RELU = 3 };

template <int EPI>
__global__ __launch_bounds__(32)
void gemm_bf16_wmma(const __bf16* __restrict__ A, const __bf16* __restrict__ BT,
                    const float* __restrict__ bias, void* __restrict__ out,
                    int M, int N, int K,
                    long long abat, long long bbat, long long obat) {
  const int lane = threadIdx.x & 31;
  const int lo   = lane & 15;
  const int hi   = lane >> 4;
  const int hi8  = hi * 8;
  const int hi16 = hi * 16;
  const int mblk = blockIdx.y;
  const int nblk = blockIdx.x;
  A  += abat * (long long)blockIdx.z;
  BT += bbat * (long long)blockIdx.z;

  v8f acc[2][4];
#pragma unroll
  for (int m2 = 0; m2 < 2; ++m2)
#pragma unroll
    for (int t = 0; t < 4; ++t) acc[m2][t] = vzero8();

  const __bf16* arow0 = A + (size_t)(mblk * 32 + lo) * K;
  const __bf16* arow1 = arow0 + (size_t)16 * K;
  for (int kb = 0; kb < K; kb += 32) {
    v16bf af0 = ld_afrag(arow0 + kb, hi8);
    v16bf af1 = ld_afrag(arow1 + kb, hi8);
    __builtin_prefetch(arow0 + kb + 64, 0, 1);   // global_prefetch_b8
    __builtin_prefetch(arow1 + kb + 64, 0, 1);
    v16bf bf[4];
#pragma unroll
    for (int t = 0; t < 4; ++t)
      bf[t] = ld_frag_contig(BT + (size_t)(nblk * 64 + t * 16 + lo) * K + kb + hi16);
#pragma unroll
    for (int t = 0; t < 4; ++t) {
      acc[0][t] = wmma_bf16(af0, bf[t], acc[0][t]);
      acc[1][t] = wmma_bf16(af1, bf[t], acc[1][t]);
    }
  }

#pragma unroll
  for (int m2 = 0; m2 < 2; ++m2) {
#pragma unroll
    for (int t = 0; t < 4; ++t) {
#pragma unroll
      for (int r = 0; r < 8; ++r) {
        const int Mi = mblk * 32 + m2 * 16 + r + hi8;
        const int Ni = nblk * 64 + t * 16 + lo;
        float val = acc[m2][t][r] + (bias ? bias[Ni] : 0.0f);
        if (EPI == EPI_F32) {
          ((float*)out)[obat * (long long)blockIdx.z + (size_t)Mi * N + Ni] = val;
        } else if (EPI == EPI_HEAD) {          // bf16 [b][h][l][dh]
          const int bb = Mi >> 11, l = Mi & (LQ - 1);
          const int h = Ni >> 6, d = Ni & (DHD - 1);
          ((__bf16*)out)[(((size_t)bb * NH + h) * LQ + l) * DHD + d] = (__bf16)val;
        } else if (EPI == EPI_HEADT) {         // bf16 [b][h][dh][l]  (V transposed)
          const int bb = Mi >> 11, l = Mi & (LQ - 1);
          const int h = Ni >> 6, d = Ni & (DHD - 1);
          ((__bf16*)out)[(((size_t)bb * NH + h) * DHD + d) * LQ + l] = (__bf16)val;
        } else {                               // ReLU -> bf16 row-major
          ((__bf16*)out)[(size_t)Mi * N + Ni] = (__bf16)fmaxf(val, 0.0f);
        }
      }
    }
  }
}

// ---------------------------------------------------------------------------
// Flash attention with distance bias. One wave per (b, h, 16-query tile).
// 64 keys per iteration; row-sums computed with a WMMA against a ones matrix
// (same bf16 P as the PV product, so normalization errors cancel).
// ---------------------------------------------------------------------------
__global__ __launch_bounds__(32)
void attn_wmma(const __bf16* __restrict__ qh, const __bf16* __restrict__ kh,
               const __bf16* __restrict__ vth, const float* __restrict__ bias,
               __bf16* __restrict__ attnh) {
  const int lane = threadIdx.x & 31;
  const int lo   = lane & 15;
  const int hi   = lane >> 4;
  const int hi8  = hi * 8;
  const int hi16 = hi * 16;
  const int q0 = blockIdx.x * 16;
  const int h  = blockIdx.y;
  const int b  = blockIdx.z;
  const size_t bh = (size_t)b * NH + h;

  __shared__ __align__(16) __bf16 pl[16 * 64];   // P tile: 16 queries x 64 keys

  // Q fragments (loop invariant): K-dim = DH = 64 -> two 16x32 fragments
  v16bf aq[2];
#pragma unroll
  for (int ks = 0; ks < 2; ++ks)
    aq[ks] = ld_afrag(qh + (bh * LQ + q0 + lo) * DHD + ks * 32, hi8);

  // all-ones B fragment for WMMA row sums
  v16bf ones;
#pragma unroll
  for (int e = 0; e < 16; ++e) ones[e] = (__bf16)1.0f;

  v8f o[4];
#pragma unroll
  for (int t = 0; t < 4; ++t) o[t] = vzero8();
  v8f ssum = vzero8();
  float mrow[8];
#pragma unroll
  for (int r = 0; r < 8; ++r) mrow[r] = -3.0e38f;

  for (int j0 = 0; j0 < LQ; j0 += 64) {
    // --- scores: four 16x16 tiles covering keys [j0, j0+64)
    v8f s[4];
#pragma unroll
    for (int tt = 0; tt < 4; ++tt) s[tt] = vzero8();
#pragma unroll
    for (int ks = 0; ks < 2; ++ks) {
      v16bf bk[4];
#pragma unroll
      for (int tt = 0; tt < 4; ++tt)
        bk[tt] = ld_frag_contig(kh + (bh * LQ + j0 + tt * 16 + lo) * DHD +
                                ks * 32 + hi16);
#pragma unroll
      for (int tt = 0; tt < 4; ++tt) s[tt] = wmma_bf16(aq[ks], bk[tt], s[tt]);
    }

    // --- scale + distance bias
#pragma unroll
    for (int r = 0; r < 8; ++r) {
      const size_t brow = ((size_t)b * LQ + (q0 + r + hi8)) * LQ + j0 + lo;
#pragma unroll
      for (int tt = 0; tt < 4; ++tt)
        s[tt][r] = s[tt][r] * 0.125f + bias[brow + tt * 16];
    }

    // --- online softmax: max-reduce across 4 tiles + 16 lanes, then exp
#pragma unroll
    for (int r = 0; r < 8; ++r) {
      float mx = fmaxf(fmaxf(s[0][r], s[1][r]), fmaxf(s[2][r], s[3][r]));
#pragma unroll
      for (int msk = 1; msk < 16; msk <<= 1) mx = fmaxf(mx, __shfl_xor(mx, msk, 32));
      const float mn = fmaxf(mrow[r], mx);
      const float corr = __expf(mrow[r] - mn);
      mrow[r] = mn;
      ssum[r] *= corr;
#pragma unroll
      for (int t = 0; t < 4; ++t) o[t][r] *= corr;
      const int row = r + hi8;
#pragma unroll
      for (int tt = 0; tt < 4; ++tt) {
        const float p = __expf(s[tt][r] - mn);
        pl[row * 64 + tt * 16 + lo] = (__bf16)p;   // stage P in LDS (C->A relayout)
      }
    }
    __syncthreads();

    // --- P @ V (and P @ ones for row sums), two 32-key chunks
#pragma unroll
    for (int c = 0; c < 2; ++c) {
      v16bf ap = ld_afrag(&pl[lo * 64 + c * 32], hi8);
      v16bf bv[4];
#pragma unroll
      for (int t = 0; t < 4; ++t)
        bv[t] = ld_frag_contig(vth + (bh * DHD + t * 16 + lo) * LQ + j0 +
                               c * 32 + hi16);
      ssum = wmma_bf16(ap, ones, ssum);            // rowsum into every column
#pragma unroll
      for (int t = 0; t < 4; ++t) o[t] = wmma_bf16(ap, bv[t], o[t]);
    }
    __syncthreads();
  }

  // --- normalize and write back as bf16 [b][l][h*DH+d] for the out-proj GEMM
#pragma unroll
  for (int t = 0; t < 4; ++t) {
#pragma unroll
    for (int r = 0; r < 8; ++r) {
      const float val = o[t][r] / ssum[r];
      attnh[((size_t)b * LQ + q0 + r + hi8) * DD + h * DHD + t * 16 + lo] = (__bf16)val;
    }
  }
}

// ---------------------------------------------------------------------------
// Helpers: conversions, bias rows, add+layernorm
// ---------------------------------------------------------------------------
__global__ void convert_x_kernel(const float* __restrict__ x, __bf16* __restrict__ xh,
                                 float* __restrict__ sq) {
  const int row = blockIdx.x, tid = threadIdx.x;      // 256 threads, D=512
  const size_t base = (size_t)row * DD;
  const float v0 = x[base + tid];
  const float v1 = x[base + 256 + tid];
  xh[base + tid] = (__bf16)v0;
  xh[base + 256 + tid] = (__bf16)v1;
  __shared__ float red[256];
  red[tid] = v0 * v0 + v1 * v1;
  __syncthreads();
  for (int s = 128; s > 0; s >>= 1) { if (tid < s) red[tid] += red[tid + s]; __syncthreads(); }
  if (tid == 0) sq[row] = red[0];
}

__global__ void convert_wT_kernel(const float* __restrict__ w, __bf16* __restrict__ wT,
                                  int K, int N) {
  const int idx = blockIdx.x * 256 + threadIdx.x;     // over K*N (multiple of 256)
  const int n = idx % N, k = idx / N;
  if (k < K) wT[(size_t)n * K + k] = (__bf16)w[idx];
}

// cd = sqrt(max(sqi+sqj-2G,0)); Dm = cd*mask2; bias = (rowmax(Dm)-Dm)*(i!=j)*mask2
__global__ void row_bias_kernel(const float* __restrict__ G, const float* __restrict__ sq,
                                const float* __restrict__ nm, float* __restrict__ biasOut) {
  const int b = blockIdx.y, i = blockIdx.x, tid = threadIdx.x;   // 256 threads, L=2048
  const size_t roff = ((size_t)b * LQ + i) * LQ;
  const float sqi = sq[b * LQ + i];
  const float mi  = nm[b * LQ + i];
  float dm[8];
  float lmax = 0.0f;
#pragma unroll
  for (int k = 0; k < 8; ++k) {
    const int j = tid + k * 256;
    const float g  = G[roff + j];
    const float d2 = sqi + sq[b * LQ + j] - 2.0f * g;
    const float cd = sqrtf(fmaxf(d2, 0.0f));
    dm[k] = cd * (mi * nm[b * LQ + j]);
    lmax = fmaxf(lmax, dm[k]);
  }
  __shared__ float red[256];
  red[tid] = lmax;
  __syncthreads();
  for (int s = 128; s > 0; s >>= 1) { if (tid < s) red[tid] = fmaxf(red[tid], red[tid + s]); __syncthreads(); }
  const float rmax = red[0];
#pragma unroll
  for (int k = 0; k < 8; ++k) {
    const int j = tid + k * 256;
    float v = (rmax - dm[k]) * (mi * nm[b * LQ + j]);
    if (j == i) v = 0.0f;
    biasOut[roff + j] = v;
  }
}

__global__ void addnorm_kernel(const float* __restrict__ a, const float* __restrict__ r,
                               const float* __restrict__ g, const float* __restrict__ be,
                               float* __restrict__ outf, __bf16* __restrict__ outh) {
  const int row = blockIdx.x, tid = threadIdx.x;      // 256 threads, D=512
  const size_t base = (size_t)row * DD;
  const float v0 = a[base + tid] + r[base + tid];
  const float v1 = a[base + 256 + tid] + r[base + 256 + tid];
  __shared__ float red[256];
  red[tid] = v0 + v1;
  __syncthreads();
  for (int s = 128; s > 0; s >>= 1) { if (tid < s) red[tid] += red[tid + s]; __syncthreads(); }
  const float mu = red[0] * (1.0f / (float)DD);
  __syncthreads();
  const float d0 = v0 - mu, d1 = v1 - mu;
  red[tid] = d0 * d0 + d1 * d1;
  __syncthreads();
  for (int s = 128; s > 0; s >>= 1) { if (tid < s) red[tid] += red[tid + s]; __syncthreads(); }
  const float inv = rsqrtf(red[0] * (1.0f / (float)DD) + 1e-5f);
  const float o0 = d0 * inv * g[tid] + be[tid];
  const float o1 = d1 * inv * g[256 + tid] + be[256 + tid];
  outf[base + tid] = o0;
  outf[base + 256 + tid] = o1;
  if (outh) { outh[base + tid] = (__bf16)o0; outh[base + 256 + tid] = (__bf16)o1; }
}

// ---------------------------------------------------------------------------
// Launcher
// ---------------------------------------------------------------------------
extern "C" void kernel_launch(void* const* d_in, const int* in_sizes, int n_in,
                              void* d_out, int out_size, void* d_ws, size_t ws_size,
                              hipStream_t stream) {
  (void)in_sizes; (void)n_in; (void)out_size; (void)ws_size;
  const float* x   = (const float*)d_in[0];
  const float* nm  = (const float*)d_in[1];
  const float* wq  = (const float*)d_in[2];  const float* bq = (const float*)d_in[3];
  const float* wk  = (const float*)d_in[4];  const float* bk = (const float*)d_in[5];
  const float* wv  = (const float*)d_in[6];  const float* bv = (const float*)d_in[7];
  const float* wo  = (const float*)d_in[8];  const float* bo = (const float*)d_in[9];
  const float* g1  = (const float*)d_in[10]; const float* be1 = (const float*)d_in[11];
  const float* w1  = (const float*)d_in[12]; const float* b1  = (const float*)d_in[13];
  const float* w2  = (const float*)d_in[14]; const float* b2  = (const float*)d_in[15];
  const float* g2  = (const float*)d_in[16]; const float* be2 = (const float*)d_in[17];

  char* ws = (char*)d_ws;
  size_t off = 0;
  auto carve = [&](size_t bytes) -> char* {
    char* p = ws + off;
    off += (bytes + 255) & ~(size_t)255;
    return p;
  };

  const size_t MTOT = (size_t)NB * LQ;  // 8192 rows
  __bf16* xh  = (__bf16*)carve(MTOT * DD * 2);
  float*  sq  = (float*) carve(MTOT * 4);
  __bf16* wqT = (__bf16*)carve((size_t)DD * DD * 2);
  __bf16* wkT = (__bf16*)carve((size_t)DD * DD * 2);
  __bf16* wvT = (__bf16*)carve((size_t)DD * DD * 2);
  __bf16* woT = (__bf16*)carve((size_t)DD * DD * 2);
  __bf16* w1T = (__bf16*)carve((size_t)DD * 4 * DD * 2);
  __bf16* w2T = (__bf16*)carve((size_t)4 * DD * DD * 2);
  char*   GbB = carve((size_t)NB * LQ * LQ * 4);           // 64 MiB bias buffer
  float*  Gb  = (float*)GbB;
  __bf16* qh_ = (__bf16*)carve(MTOT * DD * 2);
  __bf16* kh_ = (__bf16*)carve(MTOT * DD * 2);
  __bf16* vth = (__bf16*)carve(MTOT * DD * 2);
  float*  Xf  = (float*) carve(MTOT * DD * 4);
  __bf16* Xh  = (__bf16*)carve(MTOT * DD * 2);
  // Aliased buffers (lifetimes do not overlap with Gb / xh uses):
  __bf16* ah  = xh;                                  // attn out reuses xh (dead after QKV)
  float*  yb  = (float*)(GbB);                       // y = attn@wo, after bias is dead
  __bf16* h1h = (__bf16*)(GbB + ((size_t)16 << 20)); // ReLU hidden, 32 MiB
  float*  fb  = (float*)(GbB + ((size_t)48 << 20));  // FFN output, 16 MiB

  // 1. x -> bf16, row squared norms
  convert_x_kernel<<<dim3((unsigned)MTOT), 256, 0, stream>>>(x, xh, sq);

  // 2. weights -> bf16, transposed [N][K]
  convert_wT_kernel<<<dim3((DD * DD) / 256), 256, 0, stream>>>(wq, wqT, DD, DD);
  convert_wT_kernel<<<dim3((DD * DD) / 256), 256, 0, stream>>>(wk, wkT, DD, DD);
  convert_wT_kernel<<<dim3((DD * DD) / 256), 256, 0, stream>>>(wv, wvT, DD, DD);
  convert_wT_kernel<<<dim3((DD * DD) / 256), 256, 0, stream>>>(wo, woT, DD, DD);
  convert_wT_kernel<<<dim3((DD * 4 * DD) / 256), 256, 0, stream>>>(w1, w1T, DD, 4 * DD);
  convert_wT_kernel<<<dim3((4 * DD * DD) / 256), 256, 0, stream>>>(w2, w2T, 4 * DD, DD);

  // 3. Gram per batch: G[b] = x[b] @ x[b]^T  (WMMA)
  gemm_bf16_wmma<EPI_F32><<<dim3(LQ / 64, LQ / 32, NB), 32, 0, stream>>>(
      xh, xh, nullptr, Gb, LQ, LQ, DD,
      (long long)LQ * DD, (long long)LQ * DD, (long long)LQ * LQ);

  // 4. distance-residual bias rows (in place over G)
  row_bias_kernel<<<dim3(LQ, NB), 256, 0, stream>>>(Gb, sq, nm, Gb);

  // 5. QKV projections (WMMA) with head-layout epilogues
  gemm_bf16_wmma<EPI_HEAD><<<dim3(DD / 64, (unsigned)(MTOT / 32), 1), 32, 0, stream>>>(
      xh, wqT, bq, qh_, (int)MTOT, DD, DD, 0, 0, 0);
  gemm_bf16_wmma<EPI_HEAD><<<dim3(DD / 64, (unsigned)(MTOT / 32), 1), 32, 0, stream>>>(
      xh, wkT, bk, kh_, (int)MTOT, DD, DD, 0, 0, 0);
  gemm_bf16_wmma<EPI_HEADT><<<dim3(DD / 64, (unsigned)(MTOT / 32), 1), 32, 0, stream>>>(
      xh, wvT, bv, vth, (int)MTOT, DD, DD, 0, 0, 0);

  // 6. flash attention with bias (WMMA)
  attn_wmma<<<dim3(LQ / 16, NH, NB), 32, 0, stream>>>(qh_, kh_, vth, Gb, ah);

  // 7. output projection (WMMA)
  gemm_bf16_wmma<EPI_F32><<<dim3(DD / 64, (unsigned)(MTOT / 32), 1), 32, 0, stream>>>(
      ah, woT, bo, yb, (int)MTOT, DD, DD, 0, 0, 0);

  // 8. AddNorm 1
  addnorm_kernel<<<dim3((unsigned)MTOT), 256, 0, stream>>>(x, yb, g1, be1, Xf, Xh);

  // 9. FFN layer 1 with fused ReLU (WMMA)
  gemm_bf16_wmma<EPI_RELU><<<dim3(4 * DD / 64, (unsigned)(MTOT / 32), 1), 32, 0, stream>>>(
      Xh, w1T, b1, h1h, (int)MTOT, 4 * DD, DD, 0, 0, 0);

  // 10. FFN layer 2 (WMMA)
  gemm_bf16_wmma<EPI_F32><<<dim3(DD / 64, (unsigned)(MTOT / 32), 1), 32, 0, stream>>>(
      h1h, w2T, b2, fb, (int)MTOT, DD, 4 * DD, 0, 0, 0);

  // 11. AddNorm 2 -> final output (f32)
  addnorm_kernel<<<dim3((unsigned)MTOT), 256, 0, stream>>>(Xf, fb, g2, be2, (float*)d_out, nullptr);
}
